// DocREModel_81965155877351
// MI455X (gfx1250) — matrix operand order repaired and checked by reading, not compiled, not run on hardware
//
#include <hip/hip_runtime.h>

typedef _Float16 f16;
typedef __attribute__((ext_vector_type(16))) _Float16 v16h;
typedef __attribute__((ext_vector_type(8)))  float    v8f;

#define WMMA(a, b, c) __builtin_amdgcn_wmma_f32_16x16x32_f16( \
    false, (a), false, (b), (short)0, (c), false, false)

// ---------------------------------------------------------------- constants
constexpr int N_   = 3072;   // graph nodes
constexpr int HD_  = 768;    // hidden
constexpr int NH_  = 8;      // heads
constexpr int HID_ = 128;    // per-head layer1 out
constexpr int EMB_ = 768;
constexpr int NL_  = 97;
constexpr int NLP_ = 112;    // NL padded to multiple of 16
constexpr int P_   = 1024;
constexpr int KBL_ = 49152;  // EMB*BS

// ---------------------------------------------------------------- TDM
// Tensor Data Mover 2D tile load: global (rows x cols f16, row stride
// `stride_elems`) -> LDS (contiguous row-major).  D# packed per ISA
// CDNA5 ISA (group0: count/lds/global/type; group1:
// data_size=2B, tensor_dim = tile_dim (tile fully in-bounds), dim0 stride).
typedef __attribute__((ext_vector_type(4))) unsigned tdm_u4;
typedef __attribute__((ext_vector_type(8))) int      tdm_i8;
typedef __attribute__((ext_vector_type(4))) int      tdm_i4;

__device__ __forceinline__ void tdm_load_2d(unsigned lds_off, const void* gp,
                                            unsigned rows, unsigned cols,
                                            unsigned stride_elems) {
  unsigned long long ga = (unsigned long long)gp;
  tdm_u4 g0;
  g0[0] = 1u;                                    // count=1 (valid user D#)
  g0[1] = lds_off;                               // lds_addr (bytes)
  g0[2] = (unsigned)ga;                          // global_addr[31:0]
  g0[3] = (unsigned)((ga >> 32) & 0x01ffffffu) | (2u << 30);  // [56:32]|type=2
  tdm_i8 g1;
  g1[0] = 1 << 16;                               // wg_mask=0, data_size=2B
  g1[1] = (int)((cols & 0xffffu) << 16);         // tensor_dim0[15:0]
  g1[2] = (int)(((cols >> 16) & 0xffffu) |       // tensor_dim0[31:16]
                ((rows & 0xffffu) << 16));       // tensor_dim1[15:0]
  g1[3] = (int)(((rows >> 16) & 0xffffu) |       // tensor_dim1[31:16]
                ((cols & 0xffffu) << 16));       // tile_dim0
  g1[4] = (int)(rows & 0xffffu);                 // tile_dim1 (tile_dim2=0)
  g1[5] = (int)stride_elems;                     // tensor_dim0_stride[31:0]
  g1[6] = 0;
  g1[7] = 0;
  tdm_i4 z4 = {};
#if __clang_major__ >= 23
  tdm_i8 z8 = {};
  __builtin_amdgcn_tensor_load_to_lds(g0, g1, z4, z4, z8, 0);
#else
  __builtin_amdgcn_tensor_load_to_lds(g0, g1, z4, z4, 0);
#endif
}

__device__ __forceinline__ unsigned lds_off_of(const void* p) {
  return (unsigned)(unsigned long long)p;  // addr[31:0] = LDS offset
}

// ---------------------------------------------------------------- fragments
// A 16x32 f16 (ISA 7.12.2): lane L -> M=L&15, g=L>>4;
// VGPR v<4 : K = g*8 + v*2 + {0,1};  VGPR v>=4 : K = 16 + g*8 + (v-4)*2 + {0,1}
__device__ __forceinline__ v16h load_a_frag(const f16* A, int lda, int row,
                                            int g, int k0) {
  union { v16h v; unsigned u[8]; } r;
  const unsigned* p = (const unsigned*)(A + (long)row * lda + k0);
  int b = g * 4;
#pragma unroll
  for (int v = 0; v < 4; ++v) r.u[v] = p[b + v];
#pragma unroll
  for (int v = 0; v < 4; ++v) r.u[4 + v] = p[8 + b + v];
  return r.v;
}

// B 32x16 f16: lane L -> N = L&15, g=L>>4; element s : K = g*8 + s (s<8), 16 + g*8 + (s-8) (s>=8)
__device__ __forceinline__ v16h load_b_frag(const f16* B, int ldb, int krow,
                                            int n0) {
  union { v16h v; f16 e[16]; } r;
  int l = krow & 31, kb = krow - l, gg = l >> 4;
  const f16* p = B + (long)kb * ldb + n0 + (l & 15);
#pragma unroll
  for (int s = 0; s < 16; ++s) {
    int k = (s < 8) ? (gg * 8 + s) : (16 + gg * 8 + (s - 8));
    r.e[s] = p[(long)k * ldb];
  }
  return r.v;
}

// ---------------------------------------------------------------- conversions
__global__ void k_cvt(const float* __restrict__ s, f16* __restrict__ d, long n) {
  long i = (long)blockIdx.x * blockDim.x + threadIdx.x;
  if (i < n) d[i] = (f16)s[i];
}

__global__ void k_pad_wb(const float* __restrict__ s, f16* __restrict__ d) {
  long i = (long)blockIdx.x * blockDim.x + threadIdx.x;
  if (i >= (long)KBL_ * NLP_) return;
  int r = (int)(i / NLP_), c = (int)(i % NLP_);
  d[i] = (f16)(c < NL_ ? s[(long)r * NL_ + c] : 0.0f);
}

// ---------------------------------------------------------------- batched GEMM
// C[h] = A @ B[h]; one wave per 16x16 tile; B tiles staged by TDM into LDS
// with double buffering pipelined on TENSORcnt.
__global__ void k_gemm_f16out(const f16* __restrict__ A,
                              const f16* __restrict__ Bb, f16* __restrict__ Cb,
                              int K, int lda, int ldb, int ldc, long strideB,
                              long strideC) {
  int n0 = blockIdx.x * 16, m0 = blockIdx.y * 16, h = blockIdx.z;
  const f16* B = Bb + (long)h * strideB;
  f16* C = Cb + (long)h * strideC;
  int lane = threadIdx.x & 31, g = lane >> 4, mr = lane & 15;
  __shared__ f16 bt[2][32 * 16];
  unsigned bl0 = lds_off_of(&bt[0][0]), bl1 = lds_off_of(&bt[1][0]);
  tdm_load_2d(bl0, B + n0, 32, 16, (unsigned)ldb);
  v8f c = {};
  int nk = K / 32;
  for (int kb = 0; kb < nk; ++kb) {
    int cur = kb & 1;
    if (kb + 1 < nk) {
      tdm_load_2d(cur ? bl0 : bl1, B + (long)(kb + 1) * 32 * ldb + n0, 32, 16,
                  (unsigned)ldb);
      __builtin_amdgcn_s_wait_tensorcnt(1);  // oldest (current panel) done
      __builtin_prefetch(A + (long)(m0 + mr) * lda + (kb + 1) * 32, 0, 1);
    } else {
      __builtin_amdgcn_s_wait_tensorcnt(0);
    }
    v16h a = load_a_frag(A, lda, m0 + mr, g, kb * 32);
    v16h b = load_b_frag(bt[cur], 16, lane, 0);
    c = WMMA(a, b, c);
  }
  int col = n0 + mr;
#pragma unroll
  for (int v = 0; v < 8; ++v)
    C[(long)(m0 + v + 8 * g) * ldc + col] = (f16)c[v];
}

// ---------------------------------------------------------------- GAT pieces
// src[h][n] = h[h,n,:] . a[h,:Fo] ; dst uses a[h,Fo:]
__global__ void k_srcdst(const f16* __restrict__ hb, const float* __restrict__ a,
                         float* __restrict__ src, float* __restrict__ dst,
                         int Fo) {
  int idx = blockIdx.x * blockDim.x + threadIdx.x;  // h*N + n
  if (idx >= NH_ * N_) return;
  int h = idx / N_;
  const f16* row = hb + (long)idx * Fo;
  const float* ah = a + (long)h * 2 * Fo;
  float s = 0.f, d = 0.f;
  for (int o = 0; o < Fo; ++o) {
    float v = (float)row[o];
    s += v * ah[o];
    d += v * ah[Fo + o];
  }
  src[idx] = s;
  dst[idx] = d;
}

// masked-softmax row stats: max and 1/sum(exp).  wave per (h,i).
__global__ void k_stats(const float* __restrict__ src,
                        const float* __restrict__ dst,
                        const int* __restrict__ adj, float* __restrict__ rmax,
                        float* __restrict__ rsumi) {
  int i = blockIdx.x, h = blockIdx.y, lane = threadIdx.x;
  float si = src[h * N_ + i];
  const float* dh = dst + (long)h * N_;
  const int* arow = adj + (long)i * N_;
  float mx = -3.0e38f;
  for (int j = lane; j < N_; j += 32)
    if (arow[j] > 0) {
      float e = si + dh[j];
      e = e > 0.f ? e : 0.2f * e;
      mx = fmaxf(mx, e);
    }
#pragma unroll
  for (int o = 16; o; o >>= 1) mx = fmaxf(mx, __shfl_xor(mx, o, 32));
  float sm = 0.f;
  for (int j = lane; j < N_; j += 32)
    if (arow[j] > 0) {
      float e = si + dh[j];
      e = e > 0.f ? e : 0.2f * e;
      sm += __expf(e - mx);
    }
#pragma unroll
  for (int o = 16; o; o >>= 1) sm += __shfl_xor(sm, o, 32);
  if (lane == 0) {
    rmax[h * N_ + i] = mx;
    rsumi[h * N_ + i] = 1.0f / sm;
  }
}

// fused masked-softmax @ h.  Probability tile built on the fly (flash-style),
// B panel staged by TDM; both double-buffered, one barrier per K step.
// MODE 0: elu -> f16 concat layout [N][NH*FO];  MODE 1: f32 [NH][N][FO].
template <int FO, int WAVES, int NT, int MODE>
__global__ void k_att(const f16* __restrict__ hb, const float* __restrict__ src,
                      const float* __restrict__ dst, const int* __restrict__ adj,
                      const float* __restrict__ rmax,
                      const float* __restrict__ rsumi, f16* __restrict__ x1b,
                      float* __restrict__ o2) {
  int i0 = blockIdx.x * 16, h = blockIdx.y;
  int tid = threadIdx.x, wave = tid >> 5, lane = tid & 31;
  int g = lane >> 4, mr = lane & 15;
  __shared__ f16 pt[2][16 * 32];
  __shared__ f16 bp[2][32 * FO];
  const f16* hbh = hb + (long)h * N_ * FO;
  const float* dh = dst + (long)h * N_;
  constexpr int TPT = 512 / (WAVES * 32);
  // hoisted per-row softmax scalars (loop-invariant over K)
  float si[TPT], mx[TPT], rs[TPT];
  const int* arow[TPT];
  int kkv[TPT];
#pragma unroll
  for (int t = 0; t < TPT; ++t) {
    int e = tid + t * (WAVES * 32);
    int i = i0 + (e >> 5);
    kkv[t] = e & 31;
    si[t] = src[h * N_ + i];
    mx[t] = rmax[h * N_ + i];
    rs[t] = rsumi[h * N_ + i];
    arow[t] = adj + (long)i * N_;
  }
  auto fill = [&](int buf, int k0) {
#pragma unroll
    for (int t = 0; t < TPT; ++t) {
      int j = k0 + kkv[t];
      float p = 0.f;
      if (arow[t][j] > 0) {
        float ee = si[t] + dh[j];
        ee = ee > 0.f ? ee : 0.2f * ee;
        p = __expf(ee - mx[t]) * rs[t];
      }
      pt[buf][tid + t * (WAVES * 32)] = (f16)p;
    }
  };
  unsigned bl0 = lds_off_of(&bp[0][0]), bl1 = lds_off_of(&bp[1][0]);
  // prologue: panel 0 + p tile 0
  if (wave == 0) tdm_load_2d(bl0, hbh, 32, FO, FO);
  fill(0, 0);
  if (wave == 0) __builtin_amdgcn_s_wait_tensorcnt(0);
  __syncthreads();
  v8f acc[NT] = {};
  constexpr int NK = N_ / 32;
  for (int kb = 0; kb < NK; ++kb) {
    int cur = kb & 1;
    if (kb + 1 < NK) {
      if (wave == 0)
        tdm_load_2d(cur ? bl0 : bl1, hbh + (long)(kb + 1) * 32 * FO, 32, FO, FO);
      fill(cur ^ 1, (kb + 1) * 32);
    }
    v16h a = load_a_frag(pt[cur], 32, mr, g, 0);
#pragma unroll
    for (int t = 0; t < NT; ++t) {
      int n0 = (wave * NT + t) * 16;
      v16h b = load_b_frag(bp[cur], FO, lane, n0);
      acc[t] = WMMA(a, b, acc[t]);
    }
    if (wave == 0 && kb + 1 < NK) __builtin_amdgcn_s_wait_tensorcnt(0);
    __syncthreads();  // publishes next p tile + B panel; retires reads of cur
  }
#pragma unroll
  for (int t = 0; t < NT; ++t) {
    int n0 = (wave * NT + t) * 16, col = n0 + mr;
#pragma unroll
    for (int v = 0; v < 8; ++v) {
      int row = i0 + v + 8 * g;
      float val = acc[t][v];
      if constexpr (MODE == 0) {
        float e = val > 0.f ? val : (__expf(val) - 1.0f);  // elu
        x1b[(long)row * (NH_ * FO) + h * FO + col] = (f16)e;
      } else {
        o2[((long)h * N_ + row) * FO + col] = val;
      }
    }
  }
}

// g = elu(mean over heads)
__global__ void k_mean_elu(const float* __restrict__ o2, f16* __restrict__ gb) {
  long idx = (long)blockIdx.x * blockDim.x + threadIdx.x;  // n*HD + f
  if (idx >= (long)N_ * HD_) return;
  float s = 0.f;
#pragma unroll
  for (int h = 0; h < NH_; ++h) s += o2[(long)h * N_ * HD_ + idx];
  s *= 0.125f;
  s = s > 0.f ? s : (__expf(s) - 1.0f);
  gb[idx] = (f16)s;
}

// out = tanh(g[ht[:,col]] @ W + b), f16 out.  wave per tile, gathered A rows.
__global__ void k_extract(const int* __restrict__ ht, int col,
                          const f16* __restrict__ gb, const f16* __restrict__ W,
                          const float* __restrict__ bias, f16* __restrict__ ob) {
  int n0 = blockIdx.x * 16, p0 = blockIdx.y * 16;
  int lane = threadIdx.x & 31, g = lane >> 4, mr = lane & 15;
  int arow = ht[(p0 + mr) * 2 + col];
  v8f c = {};
  for (int k0 = 0; k0 < HD_; k0 += 32) {
    v16h a = load_a_frag(gb, HD_, arow, g, k0);
    v16h b = load_b_frag(W, EMB_, k0 + lane, n0);
    c = WMMA(a, b, c);
  }
  int cc = n0 + mr;
#pragma unroll
  for (int v = 0; v < 8; ++v) {
    int pr = p0 + v + 8 * g;
    ob[(long)pr * EMB_ + cc] = (f16)tanhf(c[v] + bias[cc]);
  }
}

// group bilinear fused with final GEMM.
// K = gi*64 + j, A[p][K] = hs[p][gi] * ts[p][(gi>>6)*64 + j];
// within one 32-wide K block, gi is constant (32 | 64).
__global__ void k_bilinear(const f16* __restrict__ hsb,
                           const f16* __restrict__ tsb,
                           const f16* __restrict__ Wbp,
                           const float* __restrict__ bb,
                           float* __restrict__ out) {
  int n0 = blockIdx.x * 16, p0 = blockIdx.y * 16;
  int lane = threadIdx.x & 31, g = lane >> 4, mr = lane & 15;
  int p = p0 + mr;
  const f16* hrow = hsb + (long)p * EMB_;
  const f16* trow = tsb + (long)p * EMB_;
  v8f c = {};
  for (int k0 = 0; k0 < KBL_; k0 += 32) {
    int gi = k0 >> 6;
    int tb = ((gi >> 6) << 6) + (k0 & 63);  // ts column base for localK=0
    float hs = (float)hrow[gi];
    union { v16h v; f16 e[16]; } a;
#pragma unroll
    for (int vv = 0; vv < 8; ++vv) {
      int base = ((vv < 4) ? 0 : 16) + g * 8 + (vv & 3) * 2;
      a.e[2 * vv]     = (f16)(hs * (float)trow[tb + base]);
      a.e[2 * vv + 1] = (f16)(hs * (float)trow[tb + base + 1]);
    }
    v16h b = load_b_frag(Wbp, NLP_, k0 + lane, n0);
    c = WMMA(a.v, b, c);
  }
  int cc = n0 + mr;
  if (cc < NL_) {
#pragma unroll
    for (int v = 0; v < 8; ++v)
      out[(long)(p0 + v + 8 * g) * NL_ + cc] = c[v] + bb[cc];
  }
}

// ---------------------------------------------------------------- launch
static char* take(char*& cur, size_t bytes) {
  char* r = cur;
  cur += (bytes + 255) & ~(size_t)255;
  return r;
}

extern "C" void kernel_launch(void* const* d_in, const int* in_sizes, int n_in,
                              void* d_out, int out_size, void* d_ws,
                              size_t ws_size, hipStream_t stream) {
  const float* x   = (const float*)d_in[0];
  const int*   adj = (const int*)d_in[1];
  const int*   ht  = (const int*)d_in[2];
  const float* W1  = (const float*)d_in[3];
  const float* a1  = (const float*)d_in[4];
  const float* W2  = (const float*)d_in[5];
  const float* a2  = (const float*)d_in[6];
  const float* Wh  = (const float*)d_in[7];
  const float* bh  = (const float*)d_in[8];
  const float* Wt  = (const float*)d_in[9];
  const float* bt  = (const float*)d_in[10];
  const float* Wb  = (const float*)d_in[11];
  const float* bb  = (const float*)d_in[12];
  float* out = (float*)d_out;

  char* cur = (char*)d_ws;
  f16*   xb   = (f16*)take(cur, (size_t)N_ * HD_ * 2);
  f16*   w1b  = (f16*)take(cur, (size_t)NH_ * HD_ * HID_ * 2);
  f16*   h1b  = (f16*)take(cur, (size_t)NH_ * N_ * HID_ * 2);
  float* src1 = (float*)take(cur, (size_t)NH_ * N_ * 4);
  float* dst1 = (float*)take(cur, (size_t)NH_ * N_ * 4);
  float* max1 = (float*)take(cur, (size_t)NH_ * N_ * 4);
  float* rs1  = (float*)take(cur, (size_t)NH_ * N_ * 4);
  f16*   x1b  = (f16*)take(cur, (size_t)N_ * NH_ * HID_ * 2);
  f16*   w2b  = (f16*)take(cur, (size_t)NH_ * NH_ * HID_ * HD_ * 2);
  f16*   h2b  = (f16*)take(cur, (size_t)NH_ * N_ * HD_ * 2);
  float* src2 = (float*)take(cur, (size_t)NH_ * N_ * 4);
  float* dst2 = (float*)take(cur, (size_t)NH_ * N_ * 4);
  float* max2 = (float*)take(cur, (size_t)NH_ * N_ * 4);
  float* rs2  = (float*)take(cur, (size_t)NH_ * N_ * 4);
  float* o2   = (float*)take(cur, (size_t)NH_ * N_ * HD_ * 4);
  f16*   gb   = (f16*)take(cur, (size_t)N_ * HD_ * 2);
  f16*   whb  = (f16*)take(cur, (size_t)HD_ * EMB_ * 2);
  f16*   wtb  = (f16*)take(cur, (size_t)HD_ * EMB_ * 2);
  f16*   hsb  = (f16*)take(cur, (size_t)P_ * EMB_ * 2);
  f16*   tsb  = (f16*)take(cur, (size_t)P_ * EMB_ * 2);
  f16*   wbp  = (f16*)take(cur, (size_t)KBL_ * NLP_ * 2);

  auto cvt = [&](const float* s, f16* d, long n) {
    k_cvt<<<dim3((unsigned)((n + 255) / 256)), 256, 0, stream>>>(s, d, n);
  };
  cvt(x, xb, (long)N_ * HD_);
  cvt(W1, w1b, (long)NH_ * HD_ * HID_);
  cvt(W2, w2b, (long)NH_ * NH_ * HID_ * HD_);
  cvt(Wh, whb, (long)HD_ * EMB_);
  cvt(Wt, wtb, (long)HD_ * EMB_);
  k_pad_wb<<<dim3((unsigned)(((long)KBL_ * NLP_ + 255) / 256)), 256, 0, stream>>>(
      Wb, wbp);

  // ---- GAT layer 1 ----
  k_gemm_f16out<<<dim3(HID_ / 16, N_ / 16, NH_), 32, 0, stream>>>(
      xb, w1b, h1b, HD_, HD_, HID_, HID_, (long)HD_ * HID_, (long)N_ * HID_);
  k_srcdst<<<dim3((NH_ * N_ + 255) / 256), 256, 0, stream>>>(h1b, a1, src1,
                                                             dst1, HID_);
  k_stats<<<dim3(N_, NH_), 32, 0, stream>>>(src1, dst1, adj, max1, rs1);
  k_att<HID_, 8, 1, 0><<<dim3(N_ / 16, NH_), 256, 0, stream>>>(
      h1b, src1, dst1, adj, max1, rs1, x1b, nullptr);

  // ---- GAT layer 2 ----
  k_gemm_f16out<<<dim3(HD_ / 16, N_ / 16, NH_), 32, 0, stream>>>(
      x1b, w2b, h2b, NH_ * HID_, NH_ * HID_, HD_, HD_, (long)NH_ * HID_ * HD_,
      (long)N_ * HD_);
  k_srcdst<<<dim3((NH_ * N_ + 255) / 256), 256, 0, stream>>>(h2b, a2, src2,
                                                             dst2, HD_);
  k_stats<<<dim3(N_, NH_), 32, 0, stream>>>(src2, dst2, adj, max2, rs2);
  k_att<HD_, 16, 3, 1><<<dim3(N_ / 16, NH_), 512, 0, stream>>>(
      h2b, src2, dst2, adj, max2, rs2, nullptr, o2);
  k_mean_elu<<<dim3((unsigned)(((long)N_ * HD_ + 255) / 256)), 256, 0, stream>>>(
      o2, gb);

  // ---- head/tail extractors ----
  k_extract<<<dim3(EMB_ / 16, P_ / 16), 32, 0, stream>>>(ht, 0, gb, whb, bh,
                                                         hsb);
  k_extract<<<dim3(EMB_ / 16, P_ / 16), 32, 0, stream>>>(ht, 1, gb, wtb, bt,
                                                         tsb);

  // ---- fused group-bilinear classifier ----
  k_bilinear<<<dim3(NLP_ / 16, P_ / 16), 32, 0, stream>>>(hsb, tsb, wbp, bb,
                                                          out);
  (void)in_sizes; (void)n_in; (void)out_size; (void)ws_size;
}
